// TrajectoryAttention_86612310491445
// MI455X (gfx1250) — compile-verified, matrix operand order checked by reading, not run on hardware
//
#include <hip/hip_runtime.h>
#include <math.h>

typedef __attribute__((ext_vector_type(16))) _Float16 v16h;
typedef __attribute__((ext_vector_type(8)))  _Float16 v8h;
typedef __attribute__((ext_vector_type(8)))  float    v8f;
typedef int v4i __attribute__((vector_size(16)));   // matches builtin's pointee type

#define B_  2
#define T_  2048
#define C_  1024
#define H_  16
#define D_  64
#define M_  (B_*T_)   /* 4096 rows */

#define LDSLD 40      /* padded LDS row stride in halfs: 80B, 16B-aligned, bank-spread */

// Async global->LDS path (ASYNCcnt) if the toolchain exposes it; otherwise a
// register-staged synchronous copy (same LDS pipeline, no async overlap).
#if __has_builtin(__builtin_amdgcn_global_load_async_to_lds_b128)
#define TA_ASYNC 1
#else
#define TA_ASYNC 0
#endif

__device__ __forceinline__ void cp_b128(_Float16* dst, const _Float16* src) {
#if TA_ASYNC
  __builtin_amdgcn_global_load_async_to_lds_b128(
      (__attribute__((address_space(1))) v4i*)src,
      (__attribute__((address_space(3))) v4i*)dst, 0, 0);
#else
  *(v8h*)dst = *(const v8h*)src;
#endif
}

__device__ __forceinline__ void wait_async0() {
#if TA_ASYNC
  asm volatile("s_wait_asynccnt 0x0" ::: "memory");
#endif
}

// ---------------------------------------------------------------------------
// Fragment loaders for v_wmma_f32_16x16x32_f16 (ISA 7.12.2 16-bit A 16x32):
//   lanes 0-15 : row = lane,    elems 0..7 = K k0..+7,   8..15 = K k0+16..+23
//   lanes16-31 : row = lane-16, elems 0..7 = K k0+8..+15, 8..15 = K k0+24..+31
// Two contiguous 16-byte loads per lane (global_load_b128 / ds_load_b128).
// ---------------------------------------------------------------------------
__device__ __forceinline__ v16h load_frag16(const _Float16* base, int ld,
                                            int row0, int k0) {
  const int lane = threadIdx.x & 31;
  const int r = row0 + (lane & 15);
  const int k = k0 + ((lane & 16) ? 8 : 0);
  const _Float16* p = base + (size_t)r * (size_t)ld + k;
  v8h lo = *(const v8h*)p;
  v8h hi = *(const v8h*)(p + 16);
  v16h out;
#pragma unroll
  for (int i = 0; i < 8; ++i) { out[i] = lo[i]; out[i + 8] = hi[i]; }
  return out;
}

// LDS tile fragment: tile is [64][LDSLD] halfs, k-chunk is the whole 32 cols.
__device__ __forceinline__ v16h frag_lds(const _Float16* tile, int row0) {
  const int lane = threadIdx.x & 31;
  const _Float16* p = tile + (row0 + (lane & 15)) * LDSLD + ((lane & 16) ? 8 : 0);
  v8h lo = *(const v8h*)p;
  v8h hi = *(const v8h*)(p + 16);
  v16h out;
#pragma unroll
  for (int i = 0; i < 8; ++i) { out[i] = lo[i]; out[i + 8] = hi[i]; }
  return out;
}

__device__ __forceinline__ v8f wmma_f16(v16h a, v16h b, v8f c) {
  return __builtin_amdgcn_wmma_f32_16x16x32_f16(false, a, false, b,
                                                (short)0, c, false, false);
}

__device__ __forceinline__ float rmax16(float v) {
  v = fmaxf(v, __shfl_xor(v, 1, 16));
  v = fmaxf(v, __shfl_xor(v, 2, 16));
  v = fmaxf(v, __shfl_xor(v, 4, 16));
  v = fmaxf(v, __shfl_xor(v, 8, 16));
  return v;
}
__device__ __forceinline__ float rsum16(float v) {
  v += __shfl_xor(v, 1, 16);
  v += __shfl_xor(v, 2, 16);
  v += __shfl_xor(v, 4, 16);
  v += __shfl_xor(v, 8, 16);
  return v;
}

// ---------------------------------------------------------------------------
// fp32 -> f16 conversion (grid-stride)
// ---------------------------------------------------------------------------
__global__ void ta_cvt_f16_kernel(const float* __restrict__ in,
                                  _Float16* __restrict__ out, int n) {
  int i = blockIdx.x * blockDim.x + threadIdx.x;
  int stride = gridDim.x * blockDim.x;
  for (; i < n; i += stride) out[i] = (_Float16)in[i];
}

// ---------------------------------------------------------------------------
// Fused QKV projection + bias + RoPE (+ 1/sqrt(D) folded into Q).
// Block = 128 threads (4 waves); block tile 64(M) x 64(N = one head).
// A (x rows) and Wq/Wk/Wv tiles are double-buffered in LDS via async copies;
// DMA for step k+1 overlaps ds_load+WMMA of step k.  B-fragment reads are
// software-pipelined one fragment ahead so ds_loads overlap WMMA issue.
// Q,K out: [B,H,T,D] f16.  V out transposed: [B,H,D,T] f16.
// ---------------------------------------------------------------------------
__global__ __launch_bounds__(128)
void ta_qkv_rope_kernel(const _Float16* __restrict__ Xh,
                        const _Float16* __restrict__ Wqh, const float* __restrict__ bq,
                        const _Float16* __restrict__ Wkh, const float* __restrict__ bk,
                        const _Float16* __restrict__ Wvh, const float* __restrict__ bv,
                        _Float16* __restrict__ Qo, _Float16* __restrict__ Ko,
                        _Float16* __restrict__ Vt) {
  __shared__ __align__(16) _Float16 sA[2][64 * LDSLD];
  __shared__ __align__(16) _Float16 sB[2][3][64 * LDSLD];

  const int lane   = threadIdx.x & 31;
  const int wave   = threadIdx.x >> 5;
  const int tid    = threadIdx.x;
  const int mBlock = blockIdx.x * 64;               // block row base
  const int mBase  = mBlock + wave * 16;            // wave row base
  const int nBase  = blockIdx.y * 64;               // head * 64
  const int h      = nBase >> 6;

  const _Float16* Wsrc[3] = { Wqh, Wkh, Wvh };

  v8f zero = {};
  v8f aq[4], ak[4], av[4];
#pragma unroll
  for (int nt = 0; nt < 4; ++nt) { aq[nt] = zero; ak[nt] = zero; av[nt] = zero; }

  // Stage one K-step: 4 tiles (A, Wq, Wk, Wv) of 64x32 halfs; 8 b128/thread.
  auto stage = [&](int buf, int k0) {
#pragma unroll
    for (int j = 0; j < 8; ++j) {
      const int tt  = j >> 1;                 // tile: 0=A, 1..3=Wq/Wk/Wv (uniform/j)
      const int cc  = ((j & 1) << 7) + tid;   // chunk within tile, 0..255
      const int row = cc >> 2;                // 0..63
      const int seg = (cc & 3) * 8;           // half offset 0/8/16/24
      if (tt == 0)
        cp_b128(&sA[buf][row * LDSLD + seg],
                Xh + (size_t)(mBlock + row) * C_ + k0 + seg);
      else
        cp_b128(&sB[buf][tt - 1][row * LDSLD + seg],
                Wsrc[tt - 1] + (size_t)(nBase + row) * C_ + k0 + seg);
    }
  };

  stage(0, 0);
  int buf = 0;
  for (int k0 = 0; k0 < C_; k0 += 32) {
    wait_async0();
    __syncthreads();                          // all tiles of `buf` visible
    if (k0 + 32 < C_) stage(buf ^ 1, k0 + 32);

    v16h a = frag_lds(&sA[buf][0], wave * 16);
    // one-fragment lookahead: loads for step nt+1 issue before WMMAs of nt
    v16h b0 = frag_lds(&sB[buf][0][0], 0);
    v16h b1 = frag_lds(&sB[buf][1][0], 0);
#pragma unroll
    for (int nt = 0; nt < 4; ++nt) {
      v16h b2 = frag_lds(&sB[buf][2][0], nt * 16);
      aq[nt] = wmma_f16(a, b0, aq[nt]);
      if (nt < 3) b0 = frag_lds(&sB[buf][0][0], (nt + 1) * 16);
      ak[nt] = wmma_f16(a, b1, ak[nt]);
      if (nt < 3) b1 = frag_lds(&sB[buf][1][0], (nt + 1) * 16);
      av[nt] = wmma_f16(a, b2, av[nt]);
    }
    buf ^= 1;
  }

  // --- epilogue: bias, RoPE, scatter to per-head layouts --------------------
  const int dl      = lane & 15;
  const int halfsel = (lane & 16) ? 8 : 0;

  float invf[4];
#pragma unroll
  for (int nt = 0; nt < 4; ++nt) {
    int d = nt * 16 + dl;
    // inv_freq = 10000^(-(d mod 32)/32) ; ln(10000) = 9.210340371976184
    invf[nt] = __expf(-(float)(d & 31) * (9.210340371976184f / 32.0f));
    float qb = bq[nBase + d], kb2 = bk[nBase + d], vb = bv[nBase + d];
#pragma unroll
    for (int i = 0; i < 8; ++i) { aq[nt][i] += qb; ak[nt][i] += kb2; av[nt][i] += vb; }
  }

#pragma unroll
  for (int nt = 0; nt < 4; ++nt) {
    const int d   = nt * 16 + dl;
    const int pp  = nt ^ 2;                  // partner acc holding d +/- 32
    const float s = (nt < 2) ? -1.0f : 1.0f; // rotate_half sign
#pragma unroll
    for (int i = 0; i < 8; ++i) {
      int row = mBase + i + halfsel;
      int t   = row & (T_ - 1);
      int b   = row >> 11;                   // row / T_
      float sn, cs;
      __sincosf((float)t * invf[nt], &sn, &cs);
      float qv = (aq[nt][i] * cs + s * aq[pp][i] * sn) * 0.125f; // fold 1/sqrt(64)
      float kv = (ak[nt][i] * cs + s * ak[pp][i] * sn);
      size_t hb = (size_t)(b * H_ + h);
      Qo[(hb * T_ + t) * D_ + d] = (_Float16)qv;
      Ko[(hb * T_ + t) * D_ + d] = (_Float16)kv;
      Vt[(hb * D_ + d) * T_ + t] = (_Float16)av[nt][i];
    }
  }
}

// ---------------------------------------------------------------------------
// Causal flash attention.  Block = 128 threads (4 independent waves).
// Each wave owns a 16-query tile; iterates 32 keys/step with online softmax.
// K fragments are all loaded before the S-WMMAs; V fragments are loaded at the
// top of the iteration so their latency is hidden under the softmax phase.
// P is transposed C-layout -> A-layout through a private LDS tile; DS ops are
// in-order per wave so a bare s_wait_dscnt suffices (no block barrier — waves
// have different causal trip counts).
// ---------------------------------------------------------------------------
__global__ __launch_bounds__(128)
void ta_flash_attn_kernel(const _Float16* __restrict__ Qg,
                          const _Float16* __restrict__ Kg,
                          const _Float16* __restrict__ Vtg,
                          _Float16* __restrict__ Og /* [B*T][C] f16 */) {
  __shared__ __align__(16) _Float16 ldsP[4][16][32];
  const int lane  = threadIdx.x & 31;
  const int wave  = threadIdx.x >> 5;
  const int bh    = blockIdx.y;                 // 0 .. B*H-1
  const int qBase = blockIdx.x * 64 + wave * 16;
  const int bi    = bh / H_;
  const int hi    = bh % H_;

  const _Float16* Qh = Qg  + (size_t)bh * T_ * D_;
  const _Float16* Kh = Kg  + (size_t)bh * T_ * D_;
  const _Float16* Vh = Vtg + (size_t)bh * D_ * T_;

  const v16h aq0 = load_frag16(Qh, D_, qBase, 0);   // Q rows, d 0..31
  const v16h aq1 = load_frag16(Qh, D_, qBase, 32);  // Q rows, d 32..63

  v8f zero = {};
  v8f acc[4];
#pragma unroll
  for (int nt = 0; nt < 4; ++nt) acc[nt] = zero;
  float mrun[8], lrun[8];
#pragma unroll
  for (int i = 0; i < 8; ++i) { mrun[i] = -1e30f; lrun[i] = 0.0f; }

  const int halfsel = (lane & 16) ? 8 : 0;

  for (int kb = 0; kb < qBase + 16; kb += 32) {
    // ---- issue all K and V fragment loads up front ----
    v16h k00 = load_frag16(Kh, D_, kb,      0);
    v16h k01 = load_frag16(Kh, D_, kb,      32);
    v16h k10 = load_frag16(Kh, D_, kb + 16, 0);
    v16h k11 = load_frag16(Kh, D_, kb + 16, 32);
    v16h bv0 = load_frag16(Vh, T_, 0,  kb);   // consumed after softmax:
    v16h bv1 = load_frag16(Vh, T_, 16, kb);   // latency hidden by exp/shuffle
    v16h bv2 = load_frag16(Vh, T_, 32, kb);
    v16h bv3 = load_frag16(Vh, T_, 48, kb);

    // ---- S = Q @ K^T for 32 keys (two 16-key n-tiles) ----
    v8f s0 = zero, s1 = zero;
    s0 = wmma_f16(aq0, k00, s0);
    s0 = wmma_f16(aq1, k01, s0);
    s1 = wmma_f16(aq0, k10, s1);
    s1 = wmma_f16(aq1, k11, s1);

    if (kb + 31 > qBase) {           // diagonal chunk: causal mask
      const int klo = kb + (lane & 15);
#pragma unroll
      for (int i = 0; i < 8; ++i) {
        int qt = qBase + i + halfsel;
        if (klo      > qt) s0[i] = -1e30f;
        if (klo + 16 > qt) s1[i] = -1e30f;
      }
    }

    // ---- online softmax (row stats live per (vgpr, half-wave)) ----
    float p0[8], p1[8], sc[8];
#pragma unroll
    for (int i = 0; i < 8; ++i) {
      float mx   = rmax16(fmaxf(s0[i], s1[i]));
      float mnew = fmaxf(mrun[i], mx);
      sc[i]  = __expf(mrun[i] - mnew);
      p0[i]  = __expf(s0[i] - mnew);
      p1[i]  = __expf(s1[i] - mnew);
      float rs = rsum16(p0[i] + p1[i]);
      lrun[i] = lrun[i] * sc[i] + rs;
      mrun[i] = mnew;
    }
#pragma unroll
    for (int nt = 0; nt < 4; ++nt)
#pragma unroll
      for (int i = 0; i < 8; ++i) acc[nt][i] *= sc[i];

    // ---- transpose P (C-layout -> A-layout) via per-wave LDS tile ----
#pragma unroll
    for (int i = 0; i < 8; ++i) {
      int m = i + halfsel;
      ldsP[wave][m][lane & 15]        = (_Float16)p0[i];
      ldsP[wave][m][16 + (lane & 15)] = (_Float16)p1[i];
    }
    asm volatile("s_wait_dscnt 0x0" ::: "memory");  // DS in-order within wave
    const _Float16* lp = &ldsP[wave][lane & 15][(lane & 16) ? 8 : 0];
    v8h plo = *(const v8h*)lp;
    v8h phi = *(const v8h*)(lp + 16);
    v16h ap;
#pragma unroll
    for (int i = 0; i < 8; ++i) { ap[i] = plo[i]; ap[i + 8] = phi[i]; }

    // ---- O += P @ V  (V stored [D][T]: row d contiguous in t) ----
    acc[0] = wmma_f16(ap, bv0, acc[0]);
    acc[1] = wmma_f16(ap, bv1, acc[1]);
    acc[2] = wmma_f16(ap, bv2, acc[2]);
    acc[3] = wmma_f16(ap, bv3, acc[3]);
  }

  // ---- finalize: divide by row sums, store attn-out [B*T][C] f16 ----
  float inv[8];
#pragma unroll
  for (int i = 0; i < 8; ++i) inv[i] = 1.0f / lrun[i];
#pragma unroll
  for (int nt = 0; nt < 4; ++nt) {
    int d = nt * 16 + (lane & 15);
#pragma unroll
    for (int i = 0; i < 8; ++i) {
      int t = qBase + i + halfsel;
      Og[((size_t)(bi * T_ + t)) * C_ + hi * D_ + d] = (_Float16)(acc[nt][i] * inv[i]);
    }
  }
}

// ---------------------------------------------------------------------------
// Output projection: Y = A @ Wo^T + bo, fp32 out.  Same LDS double-buffered
// async staging as the QKV kernel (2 tiles: A and Wo -> 4 b128/thread/step),
// with one-fragment lookahead on the W reads.
// ---------------------------------------------------------------------------
__global__ __launch_bounds__(128)
void ta_oproj_kernel(const _Float16* __restrict__ Ah,
                     const _Float16* __restrict__ Woh,
                     const float* __restrict__ bo,
                     float* __restrict__ Y) {
  __shared__ __align__(16) _Float16 sA[2][64 * LDSLD];
  __shared__ __align__(16) _Float16 sW[2][64 * LDSLD];

  const int lane   = threadIdx.x & 31;
  const int wave   = threadIdx.x >> 5;
  const int tid    = threadIdx.x;
  const int mBlock = blockIdx.x * 64;
  const int mBase  = mBlock + wave * 16;
  const int nBase  = blockIdx.y * 64;

  v8f zero = {};
  v8f acc[4];
#pragma unroll
  for (int nt = 0; nt < 4; ++nt) acc[nt] = zero;

  auto stage = [&](int buf, int k0) {
#pragma unroll
    for (int j = 0; j < 4; ++j) {
      const int tt  = j >> 1;                 // 0=A, 1=Wo (uniform per j)
      const int cc  = ((j & 1) << 7) + tid;   // 0..255
      const int row = cc >> 2;
      const int seg = (cc & 3) * 8;
      if (tt == 0)
        cp_b128(&sA[buf][row * LDSLD + seg],
                Ah + (size_t)(mBlock + row) * C_ + k0 + seg);
      else
        cp_b128(&sW[buf][row * LDSLD + seg],
                Woh + (size_t)(nBase + row) * C_ + k0 + seg);
    }
  };

  stage(0, 0);
  int buf = 0;
  for (int k0 = 0; k0 < C_; k0 += 32) {
    wait_async0();
    __syncthreads();
    if (k0 + 32 < C_) stage(buf ^ 1, k0 + 32);

    v16h a = frag_lds(&sA[buf][0], wave * 16);
    v16h b = frag_lds(&sW[buf][0], 0);
#pragma unroll
    for (int nt = 0; nt < 4; ++nt) {
      v16h bn;
      if (nt < 3) bn = frag_lds(&sW[buf][0], (nt + 1) * 16);
      acc[nt] = wmma_f16(a, b, acc[nt]);
      b = bn;
    }
    buf ^= 1;
  }

  const int halfsel = (lane & 16) ? 8 : 0;
#pragma unroll
  for (int nt = 0; nt < 4; ++nt) {
    int n = nBase + nt * 16 + (lane & 15);
    float bb = bo[n];
#pragma unroll
    for (int i = 0; i < 8; ++i) {
      int m = mBase + i + halfsel;
      Y[(size_t)m * C_ + n] = acc[nt][i] + bb;
    }
  }
}

// ---------------------------------------------------------------------------
extern "C" void kernel_launch(void* const* d_in, const int* in_sizes, int n_in,
                              void* d_out, int out_size, void* d_ws, size_t ws_size,
                              hipStream_t stream) {
  (void)in_sizes; (void)n_in; (void)out_size; (void)ws_size;

  const float* x  = (const float*)d_in[0];
  const float* Wq = (const float*)d_in[1];
  const float* bq = (const float*)d_in[2];
  const float* Wk = (const float*)d_in[3];
  const float* bk = (const float*)d_in[4];
  const float* Wv = (const float*)d_in[5];
  const float* bv = (const float*)d_in[6];
  const float* Wo = (const float*)d_in[7];
  const float* bo = (const float*)d_in[8];
  float* out = (float*)d_out;

  // Workspace carve-up (all f16, all 16B-aligned): 48 MB total.
  char* ws = (char*)d_ws;
  const size_t szXC = (size_t)M_ * C_ * sizeof(_Float16);            // 8 MB
  const size_t szW  = (size_t)C_ * C_ * sizeof(_Float16);            // 2 MB
  const size_t szH  = (size_t)B_ * H_ * T_ * D_ * sizeof(_Float16);  // 8 MB
  _Float16* Xh  = (_Float16*)(ws);                 ws += szXC;
  _Float16* Wqh = (_Float16*)(ws);                 ws += szW;
  _Float16* Wkh = (_Float16*)(ws);                 ws += szW;
  _Float16* Wvh = (_Float16*)(ws);                 ws += szW;
  _Float16* Woh = (_Float16*)(ws);                 ws += szW;
  _Float16* Qh  = (_Float16*)(ws);                 ws += szH;
  _Float16* Kh  = (_Float16*)(ws);                 ws += szH;
  _Float16* Vt  = (_Float16*)(ws);                 ws += szH;
  _Float16* Ah  = (_Float16*)(ws);                 ws += szXC;

  // 1) fp32 -> f16 conversions (stream-ordered)
  ta_cvt_f16_kernel<<<512, 256, 0, stream>>>(x,  Xh,  M_ * C_);
  ta_cvt_f16_kernel<<<256, 256, 0, stream>>>(Wq, Wqh, C_ * C_);
  ta_cvt_f16_kernel<<<256, 256, 0, stream>>>(Wk, Wkh, C_ * C_);
  ta_cvt_f16_kernel<<<256, 256, 0, stream>>>(Wv, Wvh, C_ * C_);
  ta_cvt_f16_kernel<<<256, 256, 0, stream>>>(Wo, Woh, C_ * C_);

  // 2) fused QKV GEMM + bias + RoPE (+ softmax scale folded into Q)
  ta_qkv_rope_kernel<<<dim3(M_ / 64, C_ / 64), 128, 0, stream>>>(
      Xh, Wqh, bq, Wkh, bk, Wvh, bv, Qh, Kh, Vt);

  // 3) causal flash attention
  ta_flash_attn_kernel<<<dim3(T_ / 64, B_ * H_), 128, 0, stream>>>(Qh, Kh, Vt, Ah);

  // 4) output projection (fp32 result)
  ta_oproj_kernel<<<dim3(M_ / 64, C_ / 64), 128, 0, stream>>>(Ah, Woh, bo, out);
}
